// PredRANN_66133906423998
// MI455X (gfx1250) — matrix operand order repaired and checked by reading, not compiled, not run on hardware
//
#include <hip/hip_runtime.h>

// ---------------------------------------------------------------------------
// PredRANN on MI455X (gfx1250): all convs + attention as bf16 WMMA GEMMs with
// fp32 accumulation; LayerNorm/softmax/gates in fp32 VALU.  Working set
// (~155MB) fits in the 192MB global L2, so the pipeline is WMMA-bound.
// This revision removes all per-element K guards (K is guaranteed %32 via a
// zero-padded input-conv weight), blocks N by 4 per wave (1 A-fragment : 4
// WMMAs), uses branchless clamped halo loads in the conv, and native __bf16
// conversions instead of manual round-to-nearest-even bit math.
// ---------------------------------------------------------------------------

typedef __attribute__((ext_vector_type(16))) __bf16 v16bf;
typedef __attribute__((ext_vector_type(8)))  float  v8f;

__device__ __forceinline__ float sigm(float x) { return 1.0f / (1.0f + __expf(-x)); }

// ---------------------------------------------------------------------------
// Generic GEMM: D[M,N] = A[M,K] * B[K,N] (+bias[M]).  fp32 in memory, bf16
// WMMA 16x16x32 with fp32 accumulate.  Compile-time FLAGS bit0: A is [K,M]
// (transposed read).  bit1: store D transposed (D[n*ldd+m]).  NB: number of
// 16-wide N sub-tiles per wave (A fragment reused NB times).  Requires
// K % 32 == 0, M % 16 == 0, N % (16*NB) == 0.  grid.z batches images.
// ---------------------------------------------------------------------------
template <int FLAGS, int NB>
__global__ __launch_bounds__(256) void gemm_wmma_kernel(
    const float* __restrict__ A, const float* __restrict__ B,
    const float* __restrict__ bias, float* __restrict__ D,
    int M, int N, int K, int lda, int ldb, int ldd,
    long long aImg, long long bImg, long long dImg)
{
  constexpr bool ATRANS = (FLAGS & 1) != 0;
  constexpr bool DTRANS = (FLAGS & 2) != 0;
  const long long z = blockIdx.z;
  A += z * aImg; B += z * bImg; D += z * dImg;
  const int ntS = (N >> 4) / NB;
  const int wave = threadIdx.x >> 5;
  const int tile = blockIdx.x * 8 + wave;
  if (tile >= (M >> 4) * ntS) return;
  const int tm  = (tile / ntS) * 16;
  const int tn0 = (tile % ntS) * (16 * NB);
  const int lane = threadIdx.x & 31;
  const int half = lane >> 4, lr = lane & 15;
  v8f acc[NB];
#pragma unroll
  for (int j = 0; j < NB; ++j) acc[j] = (v8f){};

  for (int k0 = 0; k0 < K; k0 += 32) {
    v16bf fa;
    if (!ATRANS) {
      // K-pairs are contiguous: vectorized float2 loads.
      const float2* ap2 = (const float2*)(A + (long long)(tm + lr) * lda + k0);
#pragma unroll
      for (int v = 0; v < 8; ++v) {
        int i2 = (v < 4) ? (v + half * 4) : (8 + (v - 4) + half * 4);
        float2 a = ap2[i2];
        fa[2 * v] = (__bf16)a.x; fa[2 * v + 1] = (__bf16)a.y;
      }
    } else {
#pragma unroll
      for (int v = 0; v < 8; ++v) {
        int k = k0 + ((v < 4) ? (v * 2 + half * 8) : (16 + (v - 4) * 2 + half * 8));
        fa[2 * v]     = (__bf16)A[(long long)k       * lda + (tm + lr)];
        fa[2 * v + 1] = (__bf16)A[(long long)(k + 1) * lda + (tm + lr)];
      }
    }
#pragma unroll
    for (int j = 0; j < NB; ++j) {
      const int col = tn0 + j * 16 + lr;
      v16bf fb;
#pragma unroll
      for (int v = 0; v < 8; ++v) {
        int kb = k0 + ((v < 4) ? (v * 2 + half * 8) : (16 + (v - 4) * 2 + half * 8));
        fb[2 * v]     = (__bf16)B[(long long)kb       * ldb + col];
        fb[2 * v + 1] = (__bf16)B[(long long)(kb + 1) * ldb + col];
      }
      acc[j] = __builtin_amdgcn_wmma_f32_16x16x32_bf16(false, fa, false, fb,
                                                       (short)0, acc[j], false, false);
    }
  }
#pragma unroll
  for (int j = 0; j < NB; ++j) {
#pragma unroll
    for (int v = 0; v < 8; ++v) {
      const int m = tm + v + half * 8;
      const int n = tn0 + j * 16 + lr;
      float val = acc[j][v];
      if (bias) val += bias[m];
      if (!DTRANS) D[(long long)m * ldd + n] = val;
      else         D[(long long)n * ldd + m] = val;
    }
  }
}

// ---------------------------------------------------------------------------
// 3x3 conv (pad=1) on 16x16 images as implicit GEMM: 9 shifted taps, each a
// K-chunked bf16 WMMA GEMM over input channels.  Wave computes 16 cout x 64
// positions (weights fragment reused 4x).  Halo handled by clamped loads +
// branchless zero select.  W: [cout][cin][3][3], X: [cin][256], Y: [cout][256]
// per image.  cin % 32 == 0, cout % 16 == 0.  grid.z = batch.
// ---------------------------------------------------------------------------
__global__ __launch_bounds__(256) void conv3x3_wmma_kernel(
    const float* __restrict__ W, const float* __restrict__ X,
    const float* __restrict__ bias, float* __restrict__ Y,
    int cout, int cin, long long xImg, long long yImg)
{
  const long long z = blockIdx.z;
  X += z * xImg; Y += z * yImg;
  const int wave = threadIdx.x >> 5;
  const int tile = blockIdx.x * 8 + wave;
  if (tile >= (cout >> 4) * 4) return;
  const int tm  = (tile >> 2) * 16;
  const int tn0 = (tile & 3) * 64;
  const int lane = threadIdx.x & 31;
  const int half = lane >> 4, lr = lane & 15;
  // 16-wide image: adding 16 to p only bumps the row.
  const int p0 = tn0 + lr;
  const int px = p0 & 15, py0 = p0 >> 4;
  v8f acc[4];
#pragma unroll
  for (int j = 0; j < 4; ++j) acc[j] = (v8f){};

  for (int tap = 0; tap < 9; ++tap) {
    const int dy = tap / 3 - 1, dx = tap % 3 - 1;
    const int xx = px + dx;
    const bool xin = (xx >= 0) && (xx < 16);
    const int xxc = xx < 0 ? 0 : (xx > 15 ? 15 : xx);
    int  spc[4];
    bool inb[4];
#pragma unroll
    for (int j = 0; j < 4; ++j) {
      const int yy = py0 + j + dy;
      inb[j] = xin && (yy >= 0) && (yy < 16);
      const int yyc = yy < 0 ? 0 : (yy > 15 ? 15 : yy);
      spc[j] = yyc * 16 + xxc;
    }
    for (int k0 = 0; k0 < cin; k0 += 32) {
      v16bf fa;
      const float* wp = W + (long long)(tm + lr) * cin * 9 + tap;
#pragma unroll
      for (int v = 0; v < 8; ++v) {
        int k = k0 + ((v < 4) ? (v * 2 + half * 8) : (16 + (v - 4) * 2 + half * 8));
        fa[2 * v]     = (__bf16)wp[(long long)k       * 9];
        fa[2 * v + 1] = (__bf16)wp[(long long)(k + 1) * 9];
      }
#pragma unroll
      for (int j = 0; j < 4; ++j) {
        v16bf fb;
#pragma unroll
        for (int v = 0; v < 8; ++v) {
          int kb = k0 + ((v < 4) ? (v * 2 + half * 8) : (16 + (v - 4) * 2 + half * 8));
          float b0 = X[(long long)kb       * 256 + spc[j]];
          float b1 = X[(long long)(kb + 1) * 256 + spc[j]];
          if (!inb[j]) { b0 = 0.f; b1 = 0.f; }   // branchless cndmask
          fb[2 * v] = (__bf16)b0; fb[2 * v + 1] = (__bf16)b1;
        }
        acc[j] = __builtin_amdgcn_wmma_f32_16x16x32_bf16(false, fa, false, fb,
                                                         (short)0, acc[j], false, false);
      }
    }
  }
#pragma unroll
  for (int j = 0; j < 4; ++j) {
#pragma unroll
    for (int v = 0; v < 8; ++v) {
      const int m = tm + v + half * 8;
      float val = acc[j][v] + (bias ? bias[m] : 0.f);
      Y[(long long)m * 256 + tn0 + j * 16 + lr] = val;
    }
  }
}

// ---------------------------------------------------------------------------
// LayerNorm over (C,H,W) per sample, in place.  G/Beta are [C*256].
// ---------------------------------------------------------------------------
__global__ __launch_bounds__(256) void layernorm_kernel(
    float* __restrict__ X, const float* __restrict__ G,
    const float* __restrict__ Bt, int C, long long xImg)
{
  float* x = X + (long long)blockIdx.x * xImg;
  const int n = C * 256;
  float s = 0.f, s2 = 0.f;
  for (int i = threadIdx.x; i < n; i += 256) { float v = x[i]; s += v; s2 += v * v; }
  __shared__ float rs[256], rs2[256];
  rs[threadIdx.x] = s; rs2[threadIdx.x] = s2;
  __syncthreads();
  for (int off = 128; off > 0; off >>= 1) {
    if (threadIdx.x < off) { rs[threadIdx.x] += rs[threadIdx.x + off];
                             rs2[threadIdx.x] += rs2[threadIdx.x + off]; }
    __syncthreads();
  }
  const float mean = rs[0] / n;
  const float var  = rs2[0] / n - mean * mean;
  const float inv  = rsqrtf(var + 1e-5f);
  for (int i = threadIdx.x; i < n; i += 256)
    x[i] = (x[i] - mean) * inv * G[i] + Bt[i];
}

// ST-LSTM gate math, stage 1: c_new, m_new, mem=concat(c_new,m_new).
__global__ void gates1_kernel(const float* __restrict__ xc, const float* __restrict__ hcv,
                              const float* __restrict__ mcv, float* __restrict__ cst,
                              float* __restrict__ mst, float* __restrict__ mem)
{
  int idx = blockIdx.x * blockDim.x + threadIdx.x;
  if (idx >= 4 * 64 * 256) return;
  const int p = idx & 255, ch = (idx >> 8) & 63, b = idx >> 14;
  const float* xb = xc  + (long long)b * 448 * 256;
  const float* hb = hcv + (long long)b * 256 * 256;
  const float* mb = mcv + (long long)b * 192 * 256;
  const int cp = ch * 256 + p;
  const long long so = (long long)b * 64 * 256 + cp;
  float i_ = sigm (xb[cp]            + hb[cp]);
  float f_ = sigm (xb[ 64*256 + cp]  + hb[ 64*256 + cp]);
  float g_ = tanhf(xb[128*256 + cp]  + hb[128*256 + cp]);
  float cn = f_ * cst[so] + i_ * g_;
  cst[so] = cn;
  float ip = sigm (xb[192*256 + cp]  + mb[cp]);
  float fp = sigm (xb[256*256 + cp]  + mb[ 64*256 + cp]);
  float gp = tanhf(xb[320*256 + cp]  + mb[128*256 + cp]);
  float mn = fp * mst[so] + ip * gp;
  mst[so] = mn;
  float* memb = mem + (long long)b * 128 * 256;
  memb[cp] = cn;
  memb[64 * 256 + cp] = mn;
}

// ST-LSTM gate math, stage 2: h_new = sigmoid(ox+oh+ln(conv(mem)))*tanh(conv1x1(mem)).
__global__ void gates2_kernel(const float* __restrict__ xc, const float* __restrict__ hcv,
                              const float* __restrict__ oln, const float* __restrict__ lb,
                              float* __restrict__ hout)
{
  int idx = blockIdx.x * blockDim.x + threadIdx.x;
  if (idx >= 4 * 64 * 256) return;
  const int p = idx & 255, ch = (idx >> 8) & 63, b = idx >> 14;
  const float* xb = xc  + (long long)b * 448 * 256;
  const float* hb = hcv + (long long)b * 256 * 256;
  const int cp = ch * 256 + p;
  const long long so = (long long)b * 64 * 256 + cp;
  float o = sigm(xb[384 * 256 + cp] + hb[192 * 256 + cp] + oln[so]);
  hout[so] = o * tanhf(lb[so]);
}

// Row softmax with pre-scale, in place.  grid = rows, row stride ld, length L.
__global__ __launch_bounds__(256) void softmax_kernel(float* __restrict__ S,
                                                      int L, int ld, float scale)
{
  float* row = S + (long long)blockIdx.x * ld;
  __shared__ float red[256];
  float mx = -1e30f;
  for (int i = threadIdx.x; i < L; i += 256) mx = fmaxf(mx, row[i] * scale);
  red[threadIdx.x] = mx; __syncthreads();
  for (int off = 128; off > 0; off >>= 1) {
    if (threadIdx.x < off) red[threadIdx.x] = fmaxf(red[threadIdx.x], red[threadIdx.x + off]);
    __syncthreads();
  }
  mx = red[0]; __syncthreads();
  float s = 0.f;
  for (int i = threadIdx.x; i < L; i += 256) {
    float e = __expf(row[i] * scale - mx); row[i] = e; s += e;
  }
  red[threadIdx.x] = s; __syncthreads();
  for (int off = 128; off > 0; off >>= 1) {
    if (threadIdx.x < off) red[threadIdx.x] += red[threadIdx.x + off];
    __syncthreads();
  }
  const float inv = 1.f / red[0];
  for (int i = threadIdx.x; i < L; i += 256) row[i] *= inv;
}

// Reorder conv outputs Kt/Vt [t*B+b][256][256] into attention layouts:
// Kr[b][h][c][t*256+m] (row-major K for Q*K^T), Vr[b][h][t*256+m][c].
__global__ void reorder_kv_kernel(const float* __restrict__ Kt, const float* __restrict__ Vt,
                                  float* __restrict__ Kr, float* __restrict__ Vr,
                                  int T, int ldk)
{
  long long idx = (long long)blockIdx.x * blockDim.x + threadIdx.x;
  long long total = (long long)T * 4 * 256 * 256;
  if (idx >= total) return;
  const int m  = (int)(idx & 255);
  const int co = (int)((idx >> 8) & 255);
  const int z  = (int)(idx >> 16);
  const int b = z & 3, t = z >> 2;
  const int h = co >> 5, c = co & 31;
  const float kval = Kt[idx], vval = Vt[idx];
  Kr[(((long long)b * 8 + h) * 32 + c) * ldk + t * 256 + m] = kval;
  Vr[(((long long)b * 8 + h) * ldk + t * 256 + m) * 32 + c] = vval;
}

// cur_input = mask*inputs[t] + (1-mask)*prev_output  (prediction phase).
__global__ void compose_kernel(const float* __restrict__ inp, const float* __restrict__ mask,
                               const float* __restrict__ prev, float* __restrict__ out, int t)
{
  int idx = blockIdx.x * blockDim.x + threadIdx.x;
  if (idx >= 4 * 16 * 256) return;
  const int pp = idx & 4095, b = idx >> 12;
  float m  = mask[((long long)b * 10 + (t - 10)) * 4096 + pp];
  float xi = inp [((long long)b * 20 + t)        * 4096 + pp];
  float po = prev[((long long)b * 19 + (t - 1))  * 4096 + pp];
  out[idx] = m * xi + (1.f - m) * po;
}

// Zero-pad a [M][Ksrc] weight matrix to [M][Kdst] so every GEMM has K%32==0.
__global__ void pad_weights_kernel(const float* __restrict__ src, float* __restrict__ dst,
                                   int M, int Ksrc, int Kdst)
{
  int idx = blockIdx.x * blockDim.x + threadIdx.x;
  if (idx >= M * Kdst) return;
  const int k = idx % Kdst, m = idx / Kdst;
  dst[idx] = (k < Ksrc) ? src[m * Ksrc + k] : 0.f;
}

__global__ void copyf_kernel(const float* __restrict__ src, float* __restrict__ dst, int n)
{
  int i = blockIdx.x * blockDim.x + threadIdx.x;
  if (i < n) dst[i] = src[i];
}

__global__ void zerof_kernel(float* __restrict__ dst, long long n)
{
  long long i = (long long)blockIdx.x * blockDim.x + threadIdx.x;
  if (i < n) dst[i] = 0.f;
}

// ---------------------------------------------------------------------------
// Host orchestration
// ---------------------------------------------------------------------------
static const long long IMG = 4LL * 64 * 256;   // one [B,64,16,16] state
static const int LDK = 19 * 256;               // Tcap * N

static void launch_gemm(hipStream_t s, const float* A, const float* B, const float* bias,
                        float* D, int M, int N, int K, int lda, int ldb, int ldd,
                        long long aImg, long long bImg, long long dImg, int nimg, int flags)
{
  const int nt = N >> 4;
  const bool nb4 = (nt & 3) == 0;
  const int tiles = (M >> 4) * (nb4 ? (nt >> 2) : nt);
  dim3 grid((tiles + 7) / 8, 1, nimg);
  if (nb4) {
    if (flags == 1)
      gemm_wmma_kernel<1, 4><<<grid, 256, 0, s>>>(A, B, bias, D, M, N, K, lda, ldb, ldd, aImg, bImg, dImg);
    else if (flags == 2)
      gemm_wmma_kernel<2, 4><<<grid, 256, 0, s>>>(A, B, bias, D, M, N, K, lda, ldb, ldd, aImg, bImg, dImg);
    else
      gemm_wmma_kernel<0, 4><<<grid, 256, 0, s>>>(A, B, bias, D, M, N, K, lda, ldb, ldd, aImg, bImg, dImg);
  } else {
    if (flags == 1)
      gemm_wmma_kernel<1, 1><<<grid, 256, 0, s>>>(A, B, bias, D, M, N, K, lda, ldb, ldd, aImg, bImg, dImg);
    else if (flags == 2)
      gemm_wmma_kernel<2, 1><<<grid, 256, 0, s>>>(A, B, bias, D, M, N, K, lda, ldb, ldd, aImg, bImg, dImg);
    else
      gemm_wmma_kernel<0, 1><<<grid, 256, 0, s>>>(A, B, bias, D, M, N, K, lda, ldb, ldd, aImg, bImg, dImg);
  }
}

static void launch_conv3(hipStream_t s, const float* W, const float* X, const float* bias,
                         float* Y, int cout, int cin, long long xImg, long long yImg)
{
  const int tiles = (cout / 16) * 4;
  dim3 grid((tiles + 7) / 8, 1, 4);
  conv3x3_wmma_kernel<<<grid, 256, 0, s>>>(W, X, bias, Y, cout, cin, xImg, yImg);
}

struct Ctx {
  hipStream_t s;
  float *hid, *cst, *mst, *tin, *xc, *hc, *mc, *mem, *oln, *lbuf, *raw;
  float *Q, *Kt, *Vt, *Kr, *Vr, *Sc, *aT, *hl, *h3t, *sh, *inwp;
};

static void run_stlstm(const Ctx& c, const float* const* P, const float* x,
                       const float* hprev, float* cst, float* mst, float* hout)
{
  // P: wx bx gx bxl wh bh gh bhl wm bm gm bml wo bo go bol wl bl
  launch_conv3(c.s, P[0], x,     P[1], c.xc, 448, 64, 16384, 448 * 256);
  layernorm_kernel<<<4, 256, 0, c.s>>>(c.xc, P[2], P[3], 448, 448 * 256);
  launch_conv3(c.s, P[4], hprev, P[5], c.hc, 256, 64, 16384, 256 * 256);
  layernorm_kernel<<<4, 256, 0, c.s>>>(c.hc, P[6], P[7], 256, 256 * 256);
  launch_conv3(c.s, P[8], mst,   P[9], c.mc, 192, 64, 16384, 192 * 256);
  layernorm_kernel<<<4, 256, 0, c.s>>>(c.mc, P[10], P[11], 192, 192 * 256);
  gates1_kernel<<<256, 256, 0, c.s>>>(c.xc, c.hc, c.mc, cst, mst, c.mem);
  launch_conv3(c.s, P[12], c.mem, P[13], c.oln, 64, 128, 128 * 256, 16384);
  layernorm_kernel<<<4, 256, 0, c.s>>>(c.oln, P[14], P[15], 64, 16384);
  launch_gemm(c.s, P[16], c.mem, P[17], c.lbuf, 64, 256, 128, 128, 256, 256,
              0, 128 * 256, 16384, 4, 0);
  gates2_kernel<<<256, 256, 0, c.s>>>(c.xc, c.hc, c.oln, c.lbuf, hout);
}

static void run_trans(const Ctx& c, const float* const* P, const float* qbuf,
                      const float* kvbase, int T, float* dst)
{
  // P: wq bq wk bk wv bv wo bo
  launch_gemm(c.s, P[0], qbuf,   P[1], c.Q,  256, 256, 64, 64, 256, 256, 0, 16384, 65536, 4, 0);
  launch_gemm(c.s, P[2], kvbase, P[3], c.Kt, 256, 256, 64, 64, 256, 256, 0, 16384, 65536, T * 4, 0);
  launch_gemm(c.s, P[4], kvbase, P[5], c.Vt, 256, 256, 64, 64, 256, 256, 0, 16384, 65536, T * 4, 0);
  long long tot = (long long)T * 4 * 65536;
  reorder_kv_kernel<<<(int)((tot + 255) / 256), 256, 0, c.s>>>(c.Kt, c.Vt, c.Kr, c.Vr, T, LDK);
  const float scale = 0.17677669529663687f;   // 1/sqrt(HC=32)
  for (int b = 0; b < 4; ++b) {
    // scores[h][n][(t,m)] = sum_c Q^T K, per-head batched over grid.z
    launch_gemm(c.s, c.Q + (long long)b * 65536, c.Kr + (long long)b * 8 * 32 * LDK,
                nullptr, c.Sc, 256, T * 256, 32, 256, LDK, LDK,
                8192, 32LL * LDK, 256LL * LDK, 8, /*A^T*/1);
    softmax_kernel<<<8 * 256, 256, 0, c.s>>>(c.Sc, T * 256, LDK, scale);
    // out[h][c][n] = attn * V, stored transposed into channel-major attnT
    launch_gemm(c.s, c.Sc, c.Vr + (long long)b * 8 * LDK * 32, nullptr,
                c.aT + (long long)b * 65536, 256, 32, T * 256, LDK, 32, 256,
                256LL * LDK, (long long)LDK * 32, 8192, 8, /*D^T*/2);
  }
  launch_gemm(c.s, P[6], c.aT, P[7], dst, 64, 256, 256, 256, 256, 256, 0, 65536, 16384, 4, 0);
}

extern "C" void kernel_launch(void* const* d_in, const int* in_sizes, int n_in,
                              void* d_out_v, int out_size, void* d_ws, size_t ws_size,
                              hipStream_t stream)
{
  (void)in_sizes; (void)out_size;
  if (n_in < 118) return;
  float* ws   = (float*)d_ws;
  float* dout = (float*)d_out_v;

  long long off = 0;
  auto alloc = [&](long long n) { long long o = off; off += n; return o; };
  const long long o_hid = alloc(4LL * 20 * IMG);     // hidden history [li][t][B,64,256]
  const long long o_c   = alloc(4LL * IMG);          // temporal cell state per layer
  const long long o_m   = alloc(IMG);                // spatial memory
  const long long o_tin = alloc(19LL * IMG);         // test_inputs history
  const long long o_xc  = alloc(4LL * 448 * 256);
  const long long o_hc  = alloc(4LL * 256 * 256);
  const long long o_mc  = alloc(4LL * 192 * 256);
  const long long o_mem = alloc(4LL * 128 * 256);
  const long long o_oln = alloc(IMG);
  const long long o_l   = alloc(IMG);
  const long long o_raw = alloc(5LL * 16 * 256);     // +1 image of slack for K=32 overread
  const long long o_q   = alloc(4LL * 256 * 256);
  const long long o_kt  = alloc(19LL * 4 * 256 * 256);
  const long long o_vt  = alloc(19LL * 4 * 256 * 256);
  const long long o_kr  = alloc(4LL * 8 * 32 * LDK);
  const long long o_vr  = alloc(4LL * 8 * 32 * LDK);
  const long long o_sc  = alloc(8LL * 256 * LDK);    // per-batch score scratch
  const long long o_at  = alloc(4LL * 256 * 256);
  const long long o_hl  = alloc(4LL * IMG);          // hidden_list for spatial attention
  const long long o_h3t = alloc(IMG);
  const long long o_sh  = alloc(IMG);
  const long long o_inw = alloc(64LL * 32);          // in_w zero-padded K 16 -> 32
  if (ws_size < (size_t)off * sizeof(float)) return;

  Ctx c{stream,
        ws + o_hid, ws + o_c, ws + o_m, ws + o_tin, ws + o_xc, ws + o_hc,
        ws + o_mc, ws + o_mem, ws + o_oln, ws + o_l, ws + o_raw,
        ws + o_q, ws + o_kt, ws + o_vt, ws + o_kr, ws + o_vr, ws + o_sc,
        ws + o_at, ws + o_hl, ws + o_h3t, ws + o_sh, ws + o_inw};

  const float* inputs = (const float*)d_in[0];
  const float* maskp  = (const float*)d_in[1];
  const float* rp[4][18];
  for (int li = 0; li < 4; ++li)
    for (int j = 0; j < 18; ++j) rp[li][j] = (const float*)d_in[2 + li * 18 + j];
  const float* tp[4][8];
  for (int li = 0; li < 4; ++li)
    for (int j = 0; j < 8; ++j) tp[li][j] = (const float*)d_in[74 + li * 8 + j];
  const float* sp[8];
  for (int j = 0; j < 8; ++j) sp[j] = (const float*)d_in[106 + j];
  const float* in_w  = (const float*)d_in[114];
  const float* in_b  = (const float*)d_in[115];
  const float* out_w = (const float*)d_in[116];
  const float* out_b = (const float*)d_in[117];

  // zero initial states (ws is poisoned by harness) + pad input-conv weight
  for (int li = 0; li < 4; ++li)
    zerof_kernel<<<256, 256, 0, stream>>>(c.hid + (long long)li * 20 * IMG, IMG);
  zerof_kernel<<<1024, 256, 0, stream>>>(c.cst, 4LL * IMG);
  zerof_kernel<<<256, 256, 0, stream>>>(c.mst, IMG);
  pad_weights_kernel<<<8, 256, 0, stream>>>(in_w, c.inwp, 64, 16, 32);

  for (int t = 0; t < 19; ++t) {
    float* cur = c.tin + (long long)t * IMG;
    if (t < 10) {
      launch_gemm(stream, c.inwp, inputs + (long long)t * 4096, in_b, cur,
                  64, 256, 32, 32, 256, 256, 0, 20LL * 16 * 256, 16384, 4, 0);
    } else {
      compose_kernel<<<64, 256, 0, stream>>>(inputs, maskp, dout, c.raw, t);
      launch_gemm(stream, c.inwp, c.raw, in_b, cur,
                  64, 256, 32, 32, 256, 256, 0, 4096, 16384, 4, 0);
    }
    copyf_kernel<<<256, 256, 0, stream>>>(cur, c.hl, (int)IMG);  // hidden_list[0]

    const float* x = cur;
    if (t < 10) {
      for (int li = 0; li < 4; ++li) {
        float* hout = c.hid + ((long long)li * 20 + t + 1) * IMG;
        run_stlstm(c, rp[li], x, c.hid + ((long long)li * 20 + t) * IMG,
                   c.cst + (long long)li * IMG, c.mst, hout);
        x = hout;
      }
      for (int li = 0; li < 3; ++li)
        copyf_kernel<<<256, 256, 0, stream>>>(c.hid + ((long long)li * 20 + t + 1) * IMG,
                                              c.hl + (long long)(li + 1) * IMG, (int)IMG);
      run_trans(c, sp, c.hid + (3LL * 20 + t + 1) * IMG, c.hl, 4, c.sh);
    } else {
      for (int li = 0; li < 4; ++li) {
        float* hout = c.hid + ((long long)li * 20 + t + 1) * IMG;
        run_stlstm(c, rp[li], x, c.hid + ((long long)li * 20 + t) * IMG,
                   c.cst + (long long)li * IMG, c.mst, hout);
        const float* kvb; int T;
        if (li == 0) { kvb = c.tin; T = t + 1; }                      // test_inputs[0..t]
        else         { kvb = c.hid + (long long)(li - 1) * 20 * IMG; T = t; } // hiddens[li-1][:t]
        float* dst = (li < 3) ? (c.hl + (long long)(li + 1) * IMG) : c.h3t;
        run_trans(c, tp[li], hout, kvb, T, dst);
        x = dst;
      }
      run_trans(c, sp, c.h3t, c.hl, 4, c.sh);
    }
    launch_conv3(stream, out_w, c.sh, out_b, dout + (long long)t * 4096,
                 16, 64, 16384, 19LL * 16 * 256);
  }
}
